// QSSM_v5_35605278884500
// MI455X (gfx1250) — compile-verified, matrix operand order checked by reading, not run on hardware
//
#include <hip/hip_runtime.h>
#include <hip/hip_bf16.h>

// ---------------- static dims (match reference) ----------------
typedef _Float16 h16;
typedef __attribute__((ext_vector_type(16))) _Float16 v16h;
typedef __attribute__((ext_vector_type(8)))  _Float16 v8h;
typedef __attribute__((ext_vector_type(8)))  float    v8f;

constexpr int cB = 2, cP = 16384, cC = 128, cG = 4, cCG = 32, cN = cB * cP;
constexpr int cDZ = 2, cH = 256, cW = 256, cCELLS = cB * cDZ * cH * cW;
constexpr int cDI = 256, cDS = 16, cDTR = 8, cDCONV = 4, cL = cP, cBL = cB * cP;
constexpr int cK = 1638, cS = cG * cK, cSP = 6592;   // S=6552 padded to 6592 (mult of 64)
constexpr int cNQ = 128;
constexpr float cBNS = 0.99950037f;                  // 1/sqrt(1+1e-3)

#define DEV __device__ __forceinline__

DEV float softplusf(float x) { return (x > 20.f) ? x : log1pf(__expf(x)); }
DEV float siluf(float x)     { return x / (1.f + __expf(-x)); }
DEV float geluf(float x)     { return 0.5f * x * (1.f + erff(x * 0.70710678f)); }

// ================= WMMA fragment loaders =================
// 16-bit A/B operand layout (ISA 7.12.2): lane l -> half=l>>4, r=l&15
// elems 0..7  : K = k0 + 8*half + i        (8 contiguous halves)
// elems 8..15 : K = k0 + 16 + 8*half + i-8 (8 contiguous halves)
DEV v16h load_frag_nt(const h16* __restrict__ base, int ld, int rc0, int k0, int lane) {
  int hh = lane >> 4, r = lane & 15;
  const h16* p = base + (size_t)(rc0 + r) * ld + k0 + 8 * hh;
  union { v16h v; v8h h[2]; } f;
  f.h[0] = *(const v8h*)p;
  f.h[1] = *(const v8h*)(p + 16);
  return f.v;
}

// B fragment from an LDS-resident [64][32] f16 tile (row = n-in-tile, col = k-in-tile)
DEV v16h load_frag_lds(const h16* __restrict__ tile, int row, int lane) {
  int hh = lane >> 4;
  const h16* p = tile + row * 32 + 8 * hh;
  union { v16h v; v8h h[2]; } f;
  f.h[0] = *(const v8h*)p;
  f.h[1] = *(const v8h*)(p + 16);
  return f.v;
}

// B operand for NN gemm: B is [K, N] row-major, ldb = N
DEV v16h load_frag_nn(const h16* __restrict__ Bm, int ldb, int n0, int k0, int lane) {
  int hh = lane >> 4, r = lane & 15, n = n0 + r;
  v16h o;
#pragma unroll
  for (int i = 0; i < 16; i++) {
    int k = (i < 8) ? (k0 + 8 * hh + i) : (k0 + 16 + 8 * hh + (i - 8));
    o[i] = Bm[(size_t)k * ldb + n];
  }
  return o;
}

// async copy 16B/lane from global to LDS (ASYNCcnt-tracked, ISA 15.18.3 op 98)
DEV void async_b128(unsigned lds_off, const h16* g) {
  asm volatile("global_load_async_to_lds_b128 %0, %1, off"
               :: "v"(lds_off), "v"(g) : "memory");
}
DEV void wait_async0() { asm volatile("s_wait_asynccnt 0" ::: "memory"); }
DEV void wait_async1() { asm volatile("s_wait_asynccnt 1" ::: "memory"); }

// ---- 1-tile/wave GEMM: D = A[M,K](f16) * W[N,K]^T * alpha + bias + res ----
// act: 0 none, 1 softplus
__global__ void k_gemm_nt(const h16* __restrict__ A, const h16* __restrict__ Wt,
                          float* __restrict__ D, h16* __restrict__ D16,
                          const float* __restrict__ bias, const float* __restrict__ res,
                          int M, int N, int K, float alpha, int act) {
  int wid = blockIdx.x * (blockDim.x >> 5) + (threadIdx.x >> 5);
  int tn = N >> 4;
  if (wid >= (M >> 4) * tn) return;
  int lane = threadIdx.x & 31;
  int m0 = (wid / tn) << 4, n0 = (wid % tn) << 4;
  v8f acc = {};
  for (int k0 = 0; k0 < K; k0 += 32) {
    v16h a = load_frag_nt(A, K, m0, k0, lane);
    v16h b = load_frag_nt(Wt, K, n0, k0, lane);
    acc = __builtin_amdgcn_wmma_f32_16x16x32_f16(false, a, false, b, (short)0, acc,
                                                 false, false);
  }
  int hh = lane >> 4, n = n0 + (lane & 15);
  float bv = bias ? bias[n] : 0.f;
#pragma unroll
  for (int j = 0; j < 8; j++) {
    int m = m0 + j + 8 * hh;
    size_t off = (size_t)m * N + n;
    float v = acc[j] * alpha + bv;
    if (res) v += res[off];
    if (act == 1) v = softplusf(v);
    if (D)   D[off] = v;
    if (D16) D16[off] = (h16)v;
  }
}

// ---- 4-tile/wave GEMM: block = 8 waves covering 128M x 64N macro tile.
// W tile (64x32 f16 = 4KB) staged in double-buffered LDS via async-to-LDS,
// next tile's copy overlapped with current tile's 4 WMMAs (A frag reused 4x).
__global__ void k_gemm_nt4(const h16* __restrict__ A, const h16* __restrict__ Wt,
                           float* __restrict__ D, h16* __restrict__ D16,
                           const float* __restrict__ bias, const float* __restrict__ res,
                           int M, int N, int K, float alpha, int act) {
  __shared__ alignas(16) h16 lb[2][64 * 32];
  int nt = N >> 6;
  int mt_total = (M + 127) >> 7;
  int blk = blockIdx.x;
  if (blk >= mt_total * nt) return;
  int mt = blk / nt, nb = blk % nt;
  int w = threadIdx.x >> 5, lane = threadIdx.x & 31;
  int m0 = mt * 128 + w * 16;
  int n0 = nb * 64;
  bool live = (m0 < M);
  int m0c = live ? m0 : 0;  // dead waves compute on row 0, discard at epilogue
  // staging map: 256 threads x 16B = 4KB (row = t>>2 in [0,64), col = (t&3)*8 halves)
  int srow = threadIdx.x >> 2, scol = (threadIdx.x & 3) * 8;
  const h16* gsrc = Wt + (size_t)(n0 + srow) * K + scol;
  unsigned lds_a[2];
  lds_a[0] = (unsigned)(size_t)&lb[0][srow * 32 + scol];
  lds_a[1] = (unsigned)(size_t)&lb[1][srow * 32 + scol];
  int kt = K >> 5;
  async_b128(lds_a[0], gsrc);  // stage k-tile 0
  v8f acc[4] = {{}, {}, {}, {}};
  int r = lane & 15;
  for (int t = 0; t < kt; t++) {
    if (t + 1 < kt) {
      async_b128(lds_a[(t + 1) & 1], gsrc + (t + 1) * 32);
      wait_async1();               // current tile complete (in-order), next in flight
    } else {
      wait_async0();
    }
    __syncthreads();
    const h16* cur = lb[t & 1];
    v16h a = load_frag_nt(A, K, m0c, t * 32, lane);
#pragma unroll
    for (int ni = 0; ni < 4; ni++) {
      v16h b = load_frag_lds(cur, ni * 16 + r, lane);
      acc[ni] = __builtin_amdgcn_wmma_f32_16x16x32_f16(false, a, false, b, (short)0,
                                                       acc[ni], false, false);
    }
    __syncthreads();               // protect buffer being re-staged next iter
  }
  if (!live) return;
  int hh = lane >> 4;
#pragma unroll
  for (int ni = 0; ni < 4; ni++) {
    int n = n0 + ni * 16 + r;
    float bv = bias ? bias[n] : 0.f;
#pragma unroll
    for (int j = 0; j < 8; j++) {
      int m = m0 + j + 8 * hh;
      size_t off = (size_t)m * N + n;
      float v = acc[ni][j] * alpha + bv;
      if (res) v += res[off];
      if (act == 1) v = softplusf(v);
      if (D)   D[off] = v;
      if (D16) D16[off] = (h16)v;
    }
  }
}

// D = A[M,K](f16) * B[K,N](f16)  (NN; used for attn @ V)
__global__ void k_gemm_nn(const h16* __restrict__ A, const h16* __restrict__ Bm,
                          float* __restrict__ D, h16* __restrict__ D16,
                          int M, int N, int K) {
  int wid = blockIdx.x * (blockDim.x >> 5) + (threadIdx.x >> 5);
  int tn = N >> 4;
  if (wid >= (M >> 4) * tn) return;
  int lane = threadIdx.x & 31;
  int m0 = (wid / tn) << 4, n0 = (wid % tn) << 4;
  v8f acc = {};
  for (int k0 = 0; k0 < K; k0 += 32) {
    v16h a = load_frag_nt(A, K, m0, k0, lane);
    v16h b = load_frag_nn(Bm, N, n0, k0, lane);
    acc = __builtin_amdgcn_wmma_f32_16x16x32_f16(false, a, false, b, (short)0, acc,
                                                 false, false);
  }
  int hh = lane >> 4, n = n0 + (lane & 15);
#pragma unroll
  for (int j = 0; j < 8; j++) {
    int m = m0 + j + 8 * hh;
    size_t off = (size_t)m * N + n;
    if (D)   D[off] = acc[j];
    if (D16) D16[off] = (h16)acc[j];
  }
}

// ================= elementwise / special kernels =================
__global__ void k_lin(const int* __restrict__ coords, int* __restrict__ lin) {
  int n = blockIdx.x * blockDim.x + threadIdx.x;
  if (n >= cN) return;
  int b = coords[n * 4], z = coords[n * 4 + 1], y = coords[n * 4 + 2], x = coords[n * 4 + 3];
  lin[n] = ((b * cDZ + z) * cH + y) * cW + x;
}

__global__ void k_scatter(const float* __restrict__ feats, const int* __restrict__ lin,
                          float* __restrict__ grid, int gi) {
  int t = blockIdx.x * blockDim.x + threadIdx.x;
  if (t >= cN * cCG) return;
  int c = t & 31, n = t >> 5;
  grid[(size_t)lin[n] * cCG + c] = feats[(size_t)n * cC + gi * cCG + c];
}

__global__ void k_conv(const float* __restrict__ grid, const float* __restrict__ wgt,
                       const int* __restrict__ coords, float* __restrict__ ms, int gi,
                       int kz, int ky, int kx, int dz, int dy, int dx,
                       int pz, int py, int px) {
  int t = blockIdx.x * blockDim.x + threadIdx.x;
  if (t >= cN * cCG) return;
  int oc = t & 31, n = t >> 5;
  int b = coords[n * 4], z = coords[n * 4 + 1], y = coords[n * 4 + 2], x = coords[n * 4 + 3];
  int taps = kz * ky * kx;
  float acc = 0.f;
  for (int iz = 0; iz < kz; iz++) {
    int zz = z + iz * dz - pz; if (zz < 0 || zz >= cDZ) continue;
    for (int iy = 0; iy < ky; iy++) {
      int yy = y + iy * dy - py; if (yy < 0 || yy >= cH) continue;
      for (int ix = 0; ix < kx; ix++) {
        int xx = x + ix * dx - px; if (xx < 0 || xx >= cW) continue;
        int cell = ((b * cDZ + zz) * cH + yy) * cW + xx;
        const float* gp = grid + (size_t)cell * cCG;
        const float* wp = wgt + (size_t)oc * cCG * taps + (iz * ky + iy) * kx + ix;
        for (int ic = 0; ic < cCG; ic++) acc += gp[ic] * wp[(size_t)ic * taps];
      }
    }
  }
  ms[(size_t)n * cC + gi * cCG + oc] = geluf(acc);
}

__global__ void k_gn_res(float* __restrict__ ms, const float* __restrict__ feats,
                         const float* __restrict__ g, const float* __restrict__ bt) {
  int t = blockIdx.x * blockDim.x + threadIdx.x;
  if (t >= cN * cG) return;
  int gg = t & 3, n = t >> 2;
  size_t base = (size_t)n * cC + gg * cCG;
  float v[cCG], mu = 0.f;
  for (int c = 0; c < cCG; c++) { v[c] = ms[base + c] + feats[base + c]; mu += v[c]; }
  mu *= (1.f / cCG);
  float var = 0.f;
  for (int c = 0; c < cCG; c++) { float d = v[c] - mu; var += d * d; }
  float r = rsqrtf(var * (1.f / cCG) + 1e-5f);
  for (int c = 0; c < cCG; c++)
    ms[base + c] = (v[c] - mu) * r * g[gg * cCG + c] + bt[gg * cCG + c];
}

__global__ void k_scores(const float* __restrict__ fb, unsigned long long* __restrict__ kv) {
  int t = blockIdx.x * blockDim.x + threadIdx.x;   // (b*G+g)*P+p
  if (t >= cB * cG * cP) return;
  int p = t % cP, bg = t / cP, g = bg % cG, b = bg / cG;
  const float* src = fb + ((size_t)(b * cP + p)) * cC + g * cCG;
  float s = 0.f;
  for (int c = 0; c < cCG; c++) s += src[c];
  s *= (1.f / cCG);
  unsigned u = __float_as_uint(s);
  u = (u >> 31) ? ~u : (u | 0x80000000u);  // ascending-sortable
  u = ~u;                                  // descending via ascending sort
  kv[t] = ((unsigned long long)u << 32) | (unsigned)p;
}

__global__ void k_okv(const int* __restrict__ lin, unsigned long long* __restrict__ kv) {
  int t = blockIdx.x * blockDim.x + threadIdx.x;
  if (t >= cN) return;
  kv[t] = ((unsigned long long)(unsigned)lin[t] << 32) | (unsigned)(t % cP);
}

__global__ void k_bitonic(unsigned long long* __restrict__ kv, int len, int total,
                          int k, int j) {
  int t = blockIdx.x * blockDim.x + threadIdx.x;
  if (t >= total) return;
  int i = t % len, base = (t / len) * len;
  int ixj = i ^ j;
  if (ixj <= i) return;
  unsigned long long a = kv[base + i], b = kv[base + ixj];
  bool up = ((i & k) == 0);
  if ((a > b) == up) { kv[base + i] = b; kv[base + ixj] = a; }
}

__global__ void k_inv(const unsigned long long* __restrict__ okv, int* __restrict__ inv) {
  int t = blockIdx.x * blockDim.x + threadIdx.x;
  if (t >= cN) return;
  int b = t / cP;
  int pay = (int)(okv[t] & 0xffffffffu);
  inv[b * cP + pay] = t % cP;
}

__global__ void k_gather_seq(const float* __restrict__ fb,
                             const unsigned long long* __restrict__ okv,
                             float* __restrict__ seq) {
  size_t t = (size_t)blockIdx.x * blockDim.x + threadIdx.x;
  if (t >= (size_t)cBL * cC) return;
  int c = t % cC; size_t row = t / cC; int b = (int)(row / cL);
  int p = (int)(okv[row] & 0xffffffffu);
  seq[t] = fb[((size_t)b * cP + p) * cC + c];
}

__global__ void k_rev(const float* __restrict__ s, float* __restrict__ d) {
  size_t t = (size_t)blockIdx.x * blockDim.x + threadIdx.x;
  if (t >= (size_t)cBL * cC) return;
  int c = t % cC; size_t row = t / cC; int b = (int)(row / cL), l = (int)(row % cL);
  d[t] = s[((size_t)b * cL + (cL - 1 - l)) * cC + c];
}

__global__ void k_samp(const float* __restrict__ fb,
                       const unsigned long long* __restrict__ skv,
                       h16* __restrict__ samp16) {
  size_t t = (size_t)blockIdx.x * blockDim.x + threadIdx.x;
  if (t >= (size_t)cB * cG * cK * cC) return;
  int c = t % cC; size_t r = t / cC;
  int j = (int)(r % cK); r /= cK;
  int g = (int)(r % cG), b = (int)(r / cG);
  int p = (int)(skv[((size_t)(b * cG + g)) * cP + j] & 0xffffffffu);
  samp16[((size_t)b * cSP + g * cK + j) * cC + c] = (h16)fb[((size_t)b * cP + p) * cC + c];
}

__global__ void k_rms16(const float* __restrict__ x, const float* __restrict__ wn,
                        h16* __restrict__ o) {
  int row = blockIdx.x * blockDim.x + threadIdx.x;
  if (row >= cBL) return;
  const float* xr = x + (size_t)row * cC;
  float s = 0.f;
  for (int c = 0; c < cC; c++) { float v = xr[c]; s += v * v; }
  float r = rsqrtf(s * (1.f / cC) + 1e-5f);
  h16* orow = o + (size_t)row * cC;
  for (int c = 0; c < cC; c++) orow[c] = (h16)(xr[c] * r * wn[c]);
}

__global__ void k_dwconv(const float* __restrict__ xz, const float* __restrict__ cw,
                         const float* __restrict__ cb, float* __restrict__ xc,
                         h16* __restrict__ xc16) {
  size_t t = (size_t)blockIdx.x * blockDim.x + threadIdx.x;
  if (t >= (size_t)cBL * cDI) return;
  int d = t % cDI; size_t row = t / cDI;
  int l = (int)(row % cL);
  float acc = cb[d];
#pragma unroll
  for (int kk = 0; kk < cDCONV; kk++) {
    int off = kk - (cDCONV - 1);
    if (l + off >= 0) acc += xz[(row + off) * 512 + d] * cw[d * cDCONV + kk];
  }
  float sv = siluf(acc);
  xc[row * cDI + d] = sv;
  xc16[row * cDI + d] = (h16)sv;
}

__global__ void k_delta(const float* __restrict__ dbl, const float* __restrict__ dtw,
                        const float* __restrict__ dtb, float* __restrict__ delta) {
  size_t t = (size_t)blockIdx.x * blockDim.x + threadIdx.x;
  if (t >= (size_t)cBL * cDI) return;
  int d = t % cDI; size_t row = t / cDI;
  float acc = dtb[d];
#pragma unroll
  for (int k = 0; k < cDTR; k++) acc += dbl[row * 48 + k] * dtw[d * cDTR + k];
  delta[t] = softplusf(acc);
}

// selective scan: 1 wave handles 2 channels d, 16 state lanes each
__global__ void k_scan(const float* __restrict__ delta, const float* __restrict__ xc,
                       const float* __restrict__ dbl, const float* __restrict__ xz,
                       const float* __restrict__ A_log, const float* __restrict__ Dp,
                       h16* __restrict__ y16, int zgate) {
  int wave = blockIdx.x * (blockDim.x >> 5) + (threadIdx.x >> 5);
  if (wave >= cB * (cDI / 2)) return;
  int lane = threadIdx.x & 31;
  int b = wave / (cDI / 2);
  int d = (wave % (cDI / 2)) * 2 + (lane >> 4);
  int s = lane & 15;
  float a = -__expf(A_log[d * cDS + s]);
  float Dd = Dp[d];
  float h = 0.f;
  for (int l = 0; l < cL; l++) {
    size_t row = (size_t)b * cL + l;
    if ((l & 7) == 0 && l + 8 < cL) {   // hide latency of the serial chain
      __builtin_prefetch(&delta[(row + 8) * cDI + d], 0, 0);
      __builtin_prefetch(&xc[(row + 8) * cDI + d], 0, 0);
      __builtin_prefetch(&dbl[(row + 8) * 48 + s], 0, 0);
    }
    float dl = delta[row * cDI + d];
    float xv = xc[row * cDI + d];
    float Bt = dbl[row * 48 + 8 + s];
    float Ct = dbl[row * 48 + 24 + s];
    h = h * __expf(dl * a) + dl * xv * Bt;
    float c = h * Ct;
    c += __shfl_xor(c, 1, 16);
    c += __shfl_xor(c, 2, 16);
    c += __shfl_xor(c, 4, 16);
    c += __shfl_xor(c, 8, 16);
    if (s == 0) {
      float y = c + xv * Dd;
      if (zgate) { float z = xz[row * 512 + cDI + d]; y *= siluf(z); }
      y16[row * cDI + d] = (h16)y;
    }
  }
}

__global__ void k_ln(const float* __restrict__ src, float* __restrict__ dst,
                     const float* __restrict__ g, const float* __restrict__ bt,
                     int rev, int add) {
  int row = blockIdx.x * blockDim.x + threadIdx.x;
  if (row >= cBL) return;
  int b = row / cL, l = row % cL;
  int sl = rev ? (cL - 1 - l) : l;
  const float* xr = src + ((size_t)b * cL + sl) * cC;
  float mu = 0.f;
  for (int c = 0; c < cC; c++) mu += xr[c];
  mu *= (1.f / cC);
  float var = 0.f;
  for (int c = 0; c < cC; c++) { float d = xr[c] - mu; var += d * d; }
  float r = rsqrtf(var * (1.f / cC) + 1e-5f);
  float* orow = dst + (size_t)row * cC;
  for (int c = 0; c < cC; c++) {
    float v = (xr[c] - mu) * r * g[c] + bt[c];
    if (add) orow[c] += v; else orow[c] = v;
  }
}

__global__ void k_cvt(const float* __restrict__ s, h16* __restrict__ d, long long n) {
  long long t = (long long)blockIdx.x * blockDim.x + threadIdx.x;
  if (t < n) d[t] = (h16)s[t];
}

__global__ void k_w2h(const float* __restrict__ src, h16* __restrict__ dst,
                      int on, int in_, int op, int ip) {
  int t = blockIdx.x * blockDim.x + threadIdx.x;
  if (t >= op * ip) return;
  int o = t / ip, i = t % ip;
  dst[t] = (o < on && i < in_) ? (h16)src[o * in_ + i] : (h16)0.f;
}

__global__ void k_unsort_bn(const float* __restrict__ src, const int* __restrict__ inv,
                            const float* __restrict__ g, const float* __restrict__ bt,
                            float* __restrict__ out) {
  size_t t = (size_t)blockIdx.x * blockDim.x + threadIdx.x;
  if (t >= (size_t)cN * cC) return;
  int c = t % cC; size_t n = t / cC; int b = (int)(n / cP);
  int pos = inv[n];
  float v = src[((size_t)b * cL + pos) * cC + c];
  out[t] = v * (g[c] * cBNS) + bt[c];
}

__global__ void k_softmax(const float* __restrict__ logits, h16* __restrict__ attn16) {
  __shared__ float red[256];
  int row = blockIdx.x;  // b*NQ+q
  const float* src = logits + (size_t)row * cSP;
  h16* dst = attn16 + (size_t)row * cSP;
  int t = threadIdx.x;
  float mx = -1e30f;
  for (int s = t; s < cS; s += 256) mx = fmaxf(mx, src[s]);
  red[t] = mx; __syncthreads();
  for (int o = 128; o > 0; o >>= 1) { if (t < o) red[t] = fmaxf(red[t], red[t + o]); __syncthreads(); }
  mx = red[0]; __syncthreads();
  float sm = 0.f;
  for (int s = t; s < cS; s += 256) sm += __expf(src[s] - mx);
  red[t] = sm; __syncthreads();
  for (int o = 128; o > 0; o >>= 1) { if (t < o) red[t] += red[t + o]; __syncthreads(); }
  float inv = 1.f / red[0];
  for (int s = t; s < cSP; s += 256)
    dst[s] = (s < cS) ? (h16)(__expf(src[s] - mx) * inv) : (h16)0.f;
}

__global__ void k_gate_bn(const float* __restrict__ o, const float* __restrict__ q,
                          const float* __restrict__ wg, const float* __restrict__ bg,
                          const float* __restrict__ g, const float* __restrict__ bt,
                          float* __restrict__ out) {
  int row = blockIdx.x * blockDim.x + threadIdx.x;
  if (row >= cB * cNQ) return;
  const float* orow = o + (size_t)row * cC;
  float acc = bg[0];
  for (int c = 0; c < cC; c++) acc += orow[c] * wg[c];
  float gate = 1.f / (1.f + __expf(-acc));
  for (int c = 0; c < cC; c++) {
    float qo = q[(size_t)row * cC + c] + gate * orow[c];
    out[(size_t)row * cC + c] = qo * (g[c] * cBNS) + bt[c];
  }
}

// ================= host orchestration =================
static inline int nblk(long long n, int t) { return (int)((n + t - 1) / t); }

extern "C" void kernel_launch(void* const* d_in, const int* in_sizes, int n_in,
                              void* d_out, int out_size, void* d_ws, size_t ws_size,
                              hipStream_t stream) {
  // ---- inputs (setup_inputs dict order, nested dicts flattened) ----
  const float* feats   = (const float*)d_in[0];
  const float* queries = (const float*)d_in[1];
  const float* gn_g = (const float*)d_in[2];
  const float* gn_b = (const float*)d_in[3];
  const float* convw[4] = {(const float*)d_in[4], (const float*)d_in[5],
                           (const float*)d_in[6], (const float*)d_in[7]};
  const float *m_norm[3], *m_inp[3], *m_cw[3], *m_cb[3], *m_xp[3], *m_dtw[3],
              *m_dtb[3], *m_Al[3], *m_D[3], *m_ow[3];
  for (int mi = 0; mi < 3; mi++) {
    int ib = 8 + mi * 10;
    m_norm[mi] = (const float*)d_in[ib + 0];
    m_inp[mi]  = (const float*)d_in[ib + 1];
    m_cw[mi]   = (const float*)d_in[ib + 2];
    m_cb[mi]   = (const float*)d_in[ib + 3];
    m_xp[mi]   = (const float*)d_in[ib + 4];
    m_dtw[mi]  = (const float*)d_in[ib + 5];
    m_dtb[mi]  = (const float*)d_in[ib + 6];
    m_Al[mi]   = (const float*)d_in[ib + 7];
    m_D[mi]    = (const float*)d_in[ib + 8];
    m_ow[mi]   = (const float*)d_in[ib + 9];
  }
  const float* wc   = (const float*)d_in[38]; const float* bc   = (const float*)d_in[39];
  const float* lni_g = (const float*)d_in[40]; const float* lni_b = (const float*)d_in[41];
  const float* lnj_g = (const float*)d_in[42]; const float* lnj_b = (const float*)d_in[43];
  const float* lnf_g = (const float*)d_in[44]; const float* lnf_b = (const float*)d_in[45];
  const float* fbn_g = (const float*)d_in[46]; const float* fbn_b = (const float*)d_in[47];
  const float* wq = (const float*)d_in[48]; const float* bq = (const float*)d_in[49];
  const float* wk = (const float*)d_in[50]; const float* bk = (const float*)d_in[51];
  const float* wv = (const float*)d_in[52]; const float* bv = (const float*)d_in[53];
  const float* wo = (const float*)d_in[54]; const float* bo = (const float*)d_in[55];
  const float* wg = (const float*)d_in[56]; const float* bg = (const float*)d_in[57];
  const float* abn_g = (const float*)d_in[58]; const float* abn_b = (const float*)d_in[59];
  const int* coords = (const int*)d_in[60];
  float* out = (float*)d_out;

  // ---- workspace bump allocator (~350 MB) ----
  char* wsb = (char*)d_ws;
  size_t off = 0;
  auto alloc = [&](size_t bytes) -> void* {
    void* p = wsb + off;
    off += (bytes + 255) & ~(size_t)255;
    return p;
  };
  int*  lin  = (int*)alloc((size_t)cN * 4);
  float* grid = (float*)alloc((size_t)cCELLS * cCG * 4);
  float* ms   = (float*)alloc((size_t)cN * cC * 4);             // conv out -> feats2
  unsigned long long* skv = (unsigned long long*)alloc((size_t)cB * cG * cP * 8);
  unsigned long long* okv = (unsigned long long*)alloc((size_t)cN * 8);
  int*  inv  = (int*)alloc((size_t)cN * 4);
  float* seq  = (float*)alloc((size_t)cBL * cC * 4);
  float* seqr = (float*)alloc((size_t)cBL * cC * 4);
  h16*  samp16 = (h16*)alloc((size_t)cB * cSP * cC * 2);
  // mamba scratch (shared by all 3 blocks)
  h16*  xn16 = (h16*)alloc((size_t)cBL * cC * 2);
  float* xz   = (float*)alloc((size_t)cBL * 512 * 4);
  float* xc   = (float*)alloc((size_t)cBL * cDI * 4);
  h16*  xc16 = (h16*)alloc((size_t)cBL * cDI * 2);
  float* dbl  = (float*)alloc((size_t)cBL * 48 * 4);
  float* dlt  = (float*)alloc((size_t)cBL * cDI * 4);
  h16*  y16  = (h16*)alloc((size_t)cBL * cDI * 2);
  float* mb_out = (float*)alloc((size_t)cBL * cC * 4);
  float* fused_acc = (float*)alloc((size_t)cBL * cC * 4);
  h16*  fi16 = (h16*)alloc((size_t)cBL * cC * 2);
  float* fused = (float*)alloc((size_t)cBL * cC * 4);
  // f16 weights
  h16* inp16[3]; h16* xp16[3]; h16* ow16[3];
  for (int mi = 0; mi < 3; mi++) {
    inp16[mi] = (h16*)alloc((size_t)512 * 128 * 2);
    xp16[mi]  = (h16*)alloc((size_t)48 * 256 * 2);
    ow16[mi]  = (h16*)alloc((size_t)128 * 256 * 2);
  }
  h16* wc16 = (h16*)alloc((size_t)128 * 128 * 2);
  h16* wq16 = (h16*)alloc((size_t)128 * 128 * 2);
  h16* wk16 = (h16*)alloc((size_t)128 * 128 * 2);
  h16* wv16 = (h16*)alloc((size_t)128 * 128 * 2);
  h16* wo16 = (h16*)alloc((size_t)128 * 128 * 2);
  h16* q16in = (h16*)alloc((size_t)cB * cNQ * cC * 2);
  h16* q16p  = (h16*)alloc((size_t)cB * cNQ * cC * 2);
  h16* kk16  = (h16*)alloc((size_t)cB * cSP * cC * 2);
  h16* v16   = (h16*)alloc((size_t)cB * cSP * cC * 2);
  float* logits = (float*)alloc((size_t)cB * cNQ * cSP * 4);
  h16*  attn16 = (h16*)alloc((size_t)cB * cNQ * cSP * 2);
  h16*  opre16 = (h16*)alloc((size_t)cB * cNQ * cC * 2);
  float* of    = (float*)alloc((size_t)cB * cNQ * cC * 4);

  // dispatch: wide (LDS async double-buffered) kernel when N is a multiple of 64
  auto gemm_nt = [&](const h16* A, const h16* Wt, float* D, h16* D16,
                     const float* bias, const float* res, int M, int N, int K,
                     float alpha, int act) {
    if ((N & 63) == 0) {
      int blocks = ((M + 127) / 128) * (N / 64);
      k_gemm_nt4<<<blocks, 256, 0, stream>>>(A, Wt, D, D16, bias, res, M, N, K,
                                             alpha, act);
    } else {
      int tiles = (M / 16) * (N / 16);
      k_gemm_nt<<<nblk(tiles, 8), 256, 0, stream>>>(A, Wt, D, D16, bias, res, M, N, K,
                                                    alpha, act);
    }
  };
  auto bitonic = [&](unsigned long long* kv, int len, int nseg) {
    int total = len * nseg;
    for (int k = 2; k <= len; k <<= 1)
      for (int j = k >> 1; j >= 1; j >>= 1)
        k_bitonic<<<nblk(total, 256), 256, 0, stream>>>(kv, len, total, k, j);
  };

  // ---- weight conversion to f16 ----
  for (int mi = 0; mi < 3; mi++) {
    k_w2h<<<nblk(512 * 128, 256), 256, 0, stream>>>(m_inp[mi], inp16[mi], 512, 128, 512, 128);
    k_w2h<<<nblk(48 * 256, 256), 256, 0, stream>>>(m_xp[mi], xp16[mi], 40, 256, 48, 256);
    k_w2h<<<nblk(128 * 256, 256), 256, 0, stream>>>(m_ow[mi], ow16[mi], 128, 256, 128, 256);
  }
  k_w2h<<<nblk(128 * 128, 256), 256, 0, stream>>>(wc, wc16, 128, 128, 128, 128);
  k_w2h<<<nblk(128 * 128, 256), 256, 0, stream>>>(wq, wq16, 128, 128, 128, 128);
  k_w2h<<<nblk(128 * 128, 256), 256, 0, stream>>>(wk, wk16, 128, 128, 128, 128);
  k_w2h<<<nblk(128 * 128, 256), 256, 0, stream>>>(wv, wv16, 128, 128, 128, 128);
  k_w2h<<<nblk(128 * 128, 256), 256, 0, stream>>>(wo, wo16, 128, 128, 128, 128);
  k_cvt<<<nblk((long long)cB * cNQ * cC, 256), 256, 0, stream>>>(queries, q16in,
                                                                 (long long)cB * cNQ * cC);

  // ---- multiscale sparse conv ----
  k_lin<<<nblk(cN, 256), 256, 0, stream>>>(coords, lin);
  hipMemsetAsync(grid, 0, (size_t)cCELLS * cCG * 4, stream);
  const int bp[4][9] = {{1, 3, 3, 1, 1, 1, 0, 1, 1},
                        {1, 5, 5, 1, 1, 1, 0, 2, 2},
                        {3, 3, 3, 1, 1, 1, 1, 1, 1},
                        {3, 3, 3, 1, 3, 3, 1, 3, 3}};
  for (int br = 0; br < 4; br++) {
    k_scatter<<<nblk(cN * cCG, 256), 256, 0, stream>>>(feats, lin, grid, br);
    k_conv<<<nblk(cN * cCG, 256), 256, 0, stream>>>(grid, convw[br], coords, ms, br,
        bp[br][0], bp[br][1], bp[br][2], bp[br][3], bp[br][4], bp[br][5],
        bp[br][6], bp[br][7], bp[br][8]);
  }
  k_gn_res<<<nblk(cN * cG, 256), 256, 0, stream>>>(ms, feats, gn_g, gn_b);

  // ---- top-k sampling + serialization orders ----
  k_scores<<<nblk(cB * cG * cP, 256), 256, 0, stream>>>(ms, skv);
  bitonic(skv, cP, cB * cG);
  k_okv<<<nblk(cN, 256), 256, 0, stream>>>(lin, okv);
  bitonic(okv, cP, cB);
  k_inv<<<nblk(cN, 256), 256, 0, stream>>>(okv, inv);
  hipMemsetAsync(samp16, 0, (size_t)cB * cSP * cC * 2, stream);
  k_samp<<<nblk((long long)cB * cG * cK * cC, 256), 256, 0, stream>>>(ms, skv, samp16);
  k_gather_seq<<<nblk((long long)cBL * cC, 256), 256, 0, stream>>>(ms, okv, seq);

  // ---- mamba block runner ----
  auto run_mamba = [&](const float* sin, float* sout, int mi, int zgate) {
    k_rms16<<<nblk(cBL, 256), 256, 0, stream>>>(sin, m_norm[mi], xn16);
    gemm_nt(xn16, inp16[mi], xz, nullptr, nullptr, nullptr, cBL, 512, cC, 1.f, 0);
    k_dwconv<<<nblk((long long)cBL * cDI, 256), 256, 0, stream>>>(xz, m_cw[mi], m_cb[mi],
                                                                  xc, xc16);
    gemm_nt(xc16, xp16[mi], dbl, nullptr, nullptr, nullptr, cBL, 48, cDI, 1.f, 0);
    k_delta<<<nblk((long long)cBL * cDI, 256), 256, 0, stream>>>(dbl, m_dtw[mi], m_dtb[mi], dlt);
    k_scan<<<nblk(cB * (cDI / 2), 8), 256, 0, stream>>>(dlt, xc, dbl, xz, m_Al[mi],
                                                        m_D[mi], y16, zgate);
    gemm_nt(y16, ow16[mi], sout, nullptr, nullptr, sin, cBL, cC, cDI, 1.f, 0);
  };

  // forward branch
  run_mamba(seq, mb_out, 0, 1);
  k_ln<<<nblk(cBL, 256), 256, 0, stream>>>(mb_out, fused_acc, lni_g, lni_b, 0, 0);
  // backward branch
  k_rev<<<nblk((long long)cBL * cC, 256), 256, 0, stream>>>(seq, seqr);
  run_mamba(seqr, mb_out, 1, 1);
  k_ln<<<nblk(cBL, 256), 256, 0, stream>>>(mb_out, fused_acc, lnj_g, lnj_b, 1, 1);
  // conv_fusion 1x1 + softplus
  k_cvt<<<nblk((long long)cBL * cC, 256), 256, 0, stream>>>(fused_acc, fi16,
                                                            (long long)cBL * cC);
  gemm_nt(fi16, wc16, fused, nullptr, bc, nullptr, cBL, cC, cC, 1.f, 1);
  // fusion mamba (no z gate), final LN
  run_mamba(fused, mb_out, 2, 0);
  k_ln<<<nblk(cBL, 256), 256, 0, stream>>>(mb_out, fused, lnf_g, lnf_b, 0, 0);
  // inverse permutation + BN -> first output section
  k_unsort_bn<<<nblk((long long)cN * cC, 256), 256, 0, stream>>>(fused, inv, fbn_g,
                                                                 fbn_b, out);

  // ---- query cross attention ----
  gemm_nt(q16in, wq16, nullptr, q16p, bq, nullptr, cB * cNQ, cC, cC, 1.f, 0);
  gemm_nt(samp16, wk16, nullptr, kk16, bk, nullptr, cB * cSP, cC, cC, 1.f, 0);
  gemm_nt(samp16, wv16, nullptr, v16, bv, nullptr, cB * cSP, cC, cC, 1.f, 0);
  const float scal = 0.08838834764f;  // 1/sqrt(128)
  for (int b = 0; b < cB; b++) {
    gemm_nt(q16p + (size_t)b * cNQ * cC, kk16 + (size_t)b * cSP * cC,
            logits + (size_t)b * cNQ * cSP, nullptr, nullptr, nullptr,
            cNQ, cSP, cC, scal, 0);
  }
  k_softmax<<<cB * cNQ, 256, 0, stream>>>(logits, attn16);
  for (int b = 0; b < cB; b++) {
    int tiles = (cNQ / 16) * (cC / 16);
    k_gemm_nn<<<nblk(tiles, 8), 256, 0, stream>>>(attn16 + (size_t)b * cNQ * cSP,
                                                  v16 + (size_t)b * cSP * cC,
                                                  nullptr, opre16 + (size_t)b * cNQ * cC,
                                                  cNQ, cC, cSP);
  }
  gemm_nt(opre16, wo16, of, nullptr, bo, nullptr, cB * cNQ, cC, cC, 1.f, 0);
  k_gate_bn<<<nblk(cB * cNQ, 256), 256, 0, stream>>>(of, queries, wg, bg, abn_g, abn_b,
                                                     out + (size_t)cN * cC);
  (void)in_sizes; (void)n_in; (void)out_size; (void)ws_size;
}
